// MultiHeadAttention_33792802685431
// MI455X (gfx1250) — matrix operand order repaired and checked by reading, not compiled, not run on hardware
//
#include <hip/hip_runtime.h>
#include <hip/hip_bf16.h>
#include <stdint.h>

// ---------------------------------------------------------------------------
// MI455X (gfx1250) fused multi-head attention.
// wave32, WMMA bf16 16x16x32 f32-accum, async global->LDS staging, DPP softmax.
// ---------------------------------------------------------------------------

typedef __attribute__((ext_vector_type(16))) __bf16         v16bf;
typedef __attribute__((ext_vector_type(16))) unsigned short v16us;
typedef __attribute__((ext_vector_type(8)))  float          v8f;

static constexpr int Bb = 2, Ss = 4096, Ee = 512, Hh = 8, HDd = 64;

__device__ __forceinline__ unsigned short f2bf(float f) {
  union { float f; unsigned u; } v; v.f = f;
  unsigned u = v.u;
  u += 0x7FFFu + ((u >> 16) & 1u);   // round-to-nearest-even
  return (unsigned short)(u >> 16);
}

__device__ __forceinline__ v16bf load_frag(const unsigned short* p) {
  v16us t = *(const v16us*)p;        // 32B contiguous per lane
  return __builtin_bit_cast(v16bf, t);
}

__device__ __forceinline__ v8f wmma_bf16(v16bf a, v16bf b, v8f c) {
  return __builtin_amdgcn_wmma_f32_16x16x32_bf16(false, a, false, b,
                                                 (short)0, c, false, false);
}

// Async 16-byte global->LDS copy per lane (ASYNCcnt-tracked).
// LDS dest = low 32 bits of the generic shared pointer (ISA 10.2 aperture rule).
__device__ __forceinline__ void async_copy16(const void* lds, const void* g) {
  asm volatile("global_load_async_to_lds_b128 %0, %1, off"
               :: "v"((unsigned)(uintptr_t)lds),
                  "v"((unsigned long long)(uintptr_t)g)
               : "memory");
}
__device__ __forceinline__ void wait_async0() {
  asm volatile("s_wait_asynccnt 0x0" ::: "memory");
}

// DPP 16-lane butterfly reductions (rows of 16 lanes in wave32).
template<int CTRL>
__device__ __forceinline__ float dppf(float x) {
  return __builtin_bit_cast(float, __builtin_amdgcn_update_dpp(
      0, __builtin_bit_cast(int, x), CTRL, 0xF, 0xF, true));
}
__device__ __forceinline__ float red_max16(float t) {
  t = fmaxf(t, dppf<0xB1>(t));   // quad_perm(1,0,3,2)  : xor 1
  t = fmaxf(t, dppf<0x4E>(t));   // quad_perm(2,3,0,1)  : xor 2
  t = fmaxf(t, dppf<0x141>(t));  // row_half_mirror     : fold 4-groups
  t = fmaxf(t, dppf<0x140>(t));  // row_mirror          : fold 8-groups
  return t;
}
__device__ __forceinline__ float red_sum16(float t) {
  t += dppf<0xB1>(t);
  t += dppf<0x4E>(t);
  t += dppf<0x141>(t);
  t += dppf<0x140>(t);
  return t;
}

// Fragment-order LDS placement (ISA 7.12.2, wave32):
// A 16x32 bf16: lane = m + 16*((k>>3)&1); halfword j = (k&16?8:0)+(k&7)
// B 32x16 bf16: lane = n + 16*((k>>3)&1); halfword j = (k&16?8:0)+(k&7)
// C/D 16x16 f32: element(m,n) -> vgpr v=m&7, lane = n + 16*(m>>3)

// ---------------------------------------------------------------------------
// Tiled GEMM  Y[M,N] = X[M,K] @ W[N,K]^T (+bias).  128x128 tile, K-step 32.
// ---------------------------------------------------------------------------
template<bool X_BF16, bool OUT_BF16>
__global__ __launch_bounds__(256)
void gemm_wmma_kernel(const void* __restrict__ Xv, const float* __restrict__ W,
                      const float* __restrict__ bias, void* __restrict__ Yv,
                      int M, int N, int K) {
  __shared__ __align__(32) unsigned short sA[8 * 512];
  __shared__ __align__(32) unsigned short sB[8 * 512];

  const int tid  = threadIdx.x;
  const int lane = tid & 31, wave = tid >> 5;
  const int mw = wave >> 1;        // 0..3 : 32-row strip
  const int nw = wave & 1;         // 0..1 : 64-col strip
  const int m0 = blockIdx.x * 128, n0 = blockIdx.y * 128;

  const float*          Xf = (const float*)Xv;
  const unsigned short* Xb = (const unsigned short*)Xv;

  v8f zero = {};
  v8f acc[2][4];
  for (int i = 0; i < 2; ++i)
    for (int j = 0; j < 4; ++j) acc[i][j] = zero;

  for (int k0 = 0; k0 < K; k0 += 32) {
    __syncthreads();
    // Stage A tile 128x32 in A-fragment order.
    if constexpr (X_BF16) {
      // Pure bf16 copy: 16B chunks contiguous on both sides -> async to LDS.
      for (int u = tid; u < 512; u += 256) {
        int m = u >> 2, c = u & 3;                  // c: 8-halfword chunk of k
        const unsigned short* g = Xb + (size_t)(m0 + m) * K + k0 + c * 8;
        const unsigned short* l =
            &sA[(m >> 4) * 512 + ((m & 15) + 16 * (c & 1)) * 16 + ((c >> 1) & 1) * 8];
        async_copy16(l, g);
      }
    } else {
      for (int idx = tid; idx < 128 * 32; idx += 256) {
        int m = idx >> 5, k = idx & 31;
        unsigned short h = f2bf(Xf[(size_t)(m0 + m) * K + k0 + k]);
        int la = (m & 15) + 16 * ((k >> 3) & 1);
        int j  = ((k & 16) ? 8 : 0) + (k & 7);
        sA[(m >> 4) * 512 + la * 16 + j] = h;
      }
    }
    // Stage B tile: rows n of W over k (B[k][n] = W[n][k]), f32 -> bf16.
    for (int idx = tid; idx < 128 * 32; idx += 256) {
      int n = idx >> 5, k = idx & 31;
      unsigned short h = f2bf(W[(size_t)(n0 + n) * K + k0 + k]);
      int la = (n & 15) + 16 * ((k >> 3) & 1);
      sB[(n >> 4) * 512 + la * 16 + ((k & 16) ? 8 : 0) + (k & 7)] = h;
    }
    if constexpr (X_BF16) wait_async0();
    __syncthreads();

    v16bf a[2], b[4];
    for (int i = 0; i < 2; ++i) a[i] = load_frag(&sA[(mw * 2 + i) * 512 + lane * 16]);
    for (int j = 0; j < 4; ++j) b[j] = load_frag(&sB[(nw * 4 + j) * 512 + lane * 16]);
    for (int i = 0; i < 2; ++i)
      for (int j = 0; j < 4; ++j)
        acc[i][j] = wmma_bf16(a[i], b[j], acc[i][j]);
  }

  const int hb = lane >> 4, nn = lane & 15;
  for (int i = 0; i < 2; ++i)
    for (int j = 0; j < 4; ++j)
      for (int v = 0; v < 8; ++v) {
        int row = m0 + mw * 32 + i * 16 + v + 8 * hb;
        int col = n0 + nw * 64 + j * 16 + nn;
        float val = acc[i][j][v];
        if (OUT_BF16) {
          ((unsigned short*)Yv)[(size_t)row * N + col] = f2bf(val);
        } else {
          if (bias) val += bias[col];
          ((float*)Yv)[(size_t)row * N + col] = val;
        }
      }
}

// ---------------------------------------------------------------------------
// Flash attention: grid (S/128, H, B).  Each wave owns 16 query rows.
// ---------------------------------------------------------------------------
__global__ __launch_bounds__(256)
void flash_attn_kernel(const unsigned short* __restrict__ Qb,
                       const unsigned short* __restrict__ Kb,
                       const unsigned short* __restrict__ Vb,
                       const int*            __restrict__ mask,
                       unsigned short*       __restrict__ Ctx) {
  extern __shared__ __align__(128) char smem[];
  unsigned short* lq = (unsigned short*)smem;      // 8192 halves: Q A-frags (128x64)
  unsigned short* lk = lq + 8192;                  // 4096: K^T B-frags (64 keys x 64 d)
  unsigned short* lv = lk + 4096;                  // 4096: V B-frags   (64 l x 64 d)
  unsigned short* lp = lv + 4096;                  // 8192: per-wave P A-frag scratch

  const int tid = threadIdx.x, lane = tid & 31, w = tid >> 5;
  const int q0 = blockIdx.x * 128;
  const int h  = blockIdx.y;
  const int b  = blockIdx.z;
  const int hb = lane >> 4, nn = lane & 15;
  const float SCALE = 0.044194173824159216f;       // 1/sqrt(E)

  const unsigned short* Qbase = Qb + ((size_t)b * Ss + q0) * Ee + h * HDd;
  const unsigned short* Kbase = Kb + (size_t)b * Ss * Ee + h * HDd;
  const unsigned short* Vbase = Vb + (size_t)b * Ss * Ee + h * HDd;

  // Stage Q block once: async 16B chunks directly into A-fragment order.
  for (int u = tid; u < 1024; u += 256) {
    int r = u >> 3, c = u & 7;                     // c: 8-halfword chunk of d
    const unsigned short* g = Qbase + (size_t)r * Ee + c * 8;
    const unsigned short* l =
        &lq[((r >> 4) * 2 + (c >> 2)) * 512 +
            ((r & 15) + 16 * (c & 1)) * 16 + ((c >> 1) & 1) * 8];
    async_copy16(l, g);
  }

  v8f zero = {};
  v8f acc[4];
  for (int j = 0; j < 4; ++j) acc[j] = zero;
  float m_s[8], l_s[8];
  for (int v = 0; v < 8; ++v) { m_s[v] = -__builtin_inff(); l_s[v] = 0.0f; }

  const int NKT = Ss / 64;
  for (int kt = 0; kt < NKT; ++kt) {
    __syncthreads();
    const int kbase = kt * 64;
    // Stage K^T B-frags: async 16B chunks (contiguous in global d and in frag j).
    for (int u = tid; u < 512; u += 256) {
      int kp = u >> 3;                             // key row 0..63
      int s2 = (u >> 1) & 3;                       // 16-halfword chunk of d
      int hf = u & 1;                              // 8-halfword half of chunk
      const unsigned short* g = Kbase + (size_t)(kbase + kp) * Ee + s2 * 16 + hf * 8;
      const unsigned short* l =
          &lk[((s2 >> 1) * 4 + (kp >> 4)) * 512 +
              ((kp & 15) + 16 * hf) * 16 + (s2 & 1) * 8];
      async_copy16(l, g);
    }
    // Stage V B-frags (transpose: l becomes the K-dim) via ds stores.
    for (int idx = tid; idx < 64 * 64; idx += 256) {
      int l = idx >> 6, d = idx & 63;
      unsigned short hv = Vbase[(size_t)(kbase + l) * Ee + d];
      int s = l >> 5, ll = l & 31;
      int la = (d & 15) + 16 * ((ll >> 3) & 1);
      lv[(s * 4 + (d >> 4)) * 512 + la * 16 + ((ll & 16) ? 8 : 0) + (ll & 7)] = hv;
    }
    // Prefetch next K/V tile into cache (global_prefetch_b8).
    if (kt + 1 < NKT) {
      __builtin_prefetch(Kbase + (size_t)(kbase + 64 + (tid & 63)) * Ee, 0, 1);
      __builtin_prefetch(Vbase + (size_t)(kbase + 64 + (tid & 63)) * Ee, 0, 1);
    }
    wait_async0();
    __syncthreads();

    // S = Q * K^T : 16 rows x 64 keys, 8 WMMAs.
    v8f sf[4];
    for (int g = 0; g < 4; ++g) sf[g] = zero;
    for (int s = 0; s < 2; ++s) {
      v16bf aq = load_frag(&lq[(w * 2 + s) * 512 + lane * 16]);
      for (int g = 0; g < 4; ++g) {
        v16bf bk = load_frag(&lk[(s * 4 + g) * 512 + lane * 16]);
        sf[g] = wmma_bf16(aq, bk, sf[g]);
      }
    }
    // Mask (read straight from global in C-fragment pattern) then scale.
    {
      const int* mb = mask + ((size_t)b * Ss + q0 + w * 16 + 8 * hb) * Ss + kbase + nn;
      for (int g = 0; g < 4; ++g)
        for (int v = 0; v < 8; ++v) {
          int mv = mb[(size_t)v * Ss + 16 * g];
          float sv = sf[g][v];
          sf[g][v] = mv ? sv * SCALE : -1e20f * SCALE;
        }
    }
    // Online softmax row statistics via DPP (no LDS traffic).
    float alpha[8];
    for (int v = 0; v < 8; ++v) {
      float t = fmaxf(fmaxf(sf[0][v], sf[1][v]), fmaxf(sf[2][v], sf[3][v]));
      t = red_max16(t);
      float mn = fmaxf(m_s[v], t);
      alpha[v] = __expf(m_s[v] - mn);
      m_s[v] = mn;
    }
    for (int v = 0; v < 8; ++v) {
      float t = 0.0f;
      for (int g = 0; g < 4; ++g) {
        float p = __expf(sf[g][v] - m_s[v]);
        sf[g][v] = p;
        t += p;
      }
      t = red_sum16(t);
      l_s[v] = l_s[v] * alpha[v] + t;
    }
    for (int j = 0; j < 4; ++j)
      for (int v = 0; v < 8; ++v) acc[j][v] *= alpha[v];

    // Re-layout P (C layout -> A-fragment order) through per-wave LDS scratch.
    for (int g = 0; g < 4; ++g)
      for (int v = 0; v < 8; ++v) {
        int m = v + 8 * hb;
        int n = nn + 16 * g;
        int s2 = n >> 5, nl = n & 31;
        int la = m + 16 * ((nl >> 3) & 1);
        int j  = ((nl & 16) ? 8 : 0) + (nl & 7);
        lp[w * 1024 + s2 * 512 + la * 16 + j] = f2bf(sf[g][v]);
      }
    // O += P * V : 8 WMMAs.
    for (int s = 0; s < 2; ++s) {
      v16bf ap = load_frag(&lp[w * 1024 + s * 512 + lane * 16]);
      for (int j = 0; j < 4; ++j) {
        v16bf bv = load_frag(&lv[(s * 4 + j) * 512 + lane * 16]);
        acc[j] = wmma_bf16(ap, bv, acc[j]);
      }
    }
  }

  // Normalize and write bf16 context [B,S,E].
  for (int j = 0; j < 4; ++j)
    for (int v = 0; v < 8; ++v) {
      int m = v + 8 * hb;
      float val = acc[j][v] / l_s[v];
      Ctx[((size_t)b * Ss + q0 + w * 16 + m) * Ee + h * HDd + j * 16 + nn] = f2bf(val);
    }
}

// ---------------------------------------------------------------------------
extern "C" void kernel_launch(void* const* d_in, const int* in_sizes, int n_in,
                              void* d_out, int out_size, void* d_ws, size_t ws_size,
                              hipStream_t stream) {
  const float* values = (const float*)d_in[0];
  const float* keys   = (const float*)d_in[1];
  const float* query  = (const float*)d_in[2];
  const int*   mask   = (const int*)d_in[3];
  const float* Wv = (const float*)d_in[4];
  const float* Wk = (const float*)d_in[5];
  const float* Wq = (const float*)d_in[6];
  const float* Wo = (const float*)d_in[7];
  const float* bo = (const float*)d_in[8];

  const int M = Bb * Ss;                       // 8192
  const size_t PROJ = (size_t)M * Ee;          // 4M bf16 elems per buffer
  unsigned short* Qb = (unsigned short*)d_ws;
  unsigned short* Kb = Qb + PROJ;
  unsigned short* Vb = Kb + PROJ;
  unsigned short* Cx = Vb + PROJ;

  dim3 blk(256);
  dim3 g1(M / 128, Ee / 128);                  // 64 x 4

  gemm_wmma_kernel<false, true><<<g1, blk, 0, stream>>>(
      (const void*)query, Wq, nullptr, (void*)Qb, M, Ee, Ee);
  gemm_wmma_kernel<false, true><<<g1, blk, 0, stream>>>(
      (const void*)keys, Wk, nullptr, (void*)Kb, M, Ee, Ee);
  gemm_wmma_kernel<false, true><<<g1, blk, 0, stream>>>(
      (const void*)values, Wv, nullptr, (void*)Vb, M, Ee, Ee);

  dim3 g2(Ss / 128, Hh, Bb);                   // 32 x 8 x 2
  size_t smem = (8192 + 4096 + 4096 + 8192) * sizeof(unsigned short);
  flash_attn_kernel<<<g2, blk, smem, stream>>>(Qb, Kb, Vb, mask, Cx);

  gemm_wmma_kernel<true, false><<<g1, blk, 0, stream>>>(
      (const void*)Cx, Wo, bo, d_out, M, Ee, Ee);
}
